// SelfAttention2d_6313601925671
// MI455X (gfx1250) — matrix operand order repaired and checked by reading, not compiled, not run on hardware
//
#include <hip/hip_runtime.h>
#include <hip/hip_bf16.h>

// ---------------------------------------------------------------------------
// Types for CDNA5 WMMA bf16 16x16x32 (wave32): A,B = v16bf (8 VGPRs), C/D = v8f
// ---------------------------------------------------------------------------
typedef __attribute__((ext_vector_type(16))) __bf16 v16bf;
typedef __attribute__((ext_vector_type(8)))  __bf16 v8bf;
typedef __attribute__((ext_vector_type(8)))  float  v8f;

__device__ __forceinline__ v16bf bf_combine(v8bf lo, v8bf hi) {
    v16bf r;
#pragma unroll
    for (int i = 0; i < 8; ++i) { r[i] = lo[i]; r[8 + i] = hi[i]; }
    return r;
}

__device__ __forceinline__ v8f wmma_bf16(v16bf a, v16bf b, v8f c) {
    // D = A(16x32) * B(32x16) + C, fp32 accumulate
    return __builtin_amdgcn_wmma_f32_16x16x32_bf16(false, a, false, b,
                                                   (short)0, c, false, false);
}

__device__ __forceinline__ void lds_fence() {
    // order intra-wave LDS RAW (spill -> scan) without a full barrier
    asm volatile("s_wait_dscnt 0x0" ::: "memory");
}

#define N_TOK  4096   // H*W
#define C_CH   256
#define D_CH   32
#define JTOT   320    // 32 q + 32 k + 256 v

// ---------------------------------------------------------------------------
// Kernel 1: fused QKV projection.  out[n,j] = sum_c x[c,n]*w[j,c] + b[j]
// Grid: B * (N/32) blocks, 256 threads (8 waves). Each wave: 16 rows x 80 cols.
// q -> qT[b][n][32] bf16, k -> kT[b][n][32] bf16, v -> vN[b][c][n] bf16
// ---------------------------------------------------------------------------
__global__ __launch_bounds__(256) void qkv_proj_kernel(
    const float* __restrict__ x,
    const float* __restrict__ wq, const float* __restrict__ bq,
    const float* __restrict__ wk, const float* __restrict__ bk,
    const float* __restrict__ wv, const float* __restrict__ bv,
    __bf16* __restrict__ qT, __bf16* __restrict__ kT, __bf16* __restrict__ vN)
{
    const int b     = blockIdx.x >> 7;
    const int nbase = (blockIdx.x & 127) * 32;
    const int t     = threadIdx.x;
    const int lane  = t & 31;
    const int w     = t >> 5;
    const int hi    = lane >> 4;
    const int lo    = lane & 15;

    __shared__ __align__(16) __bf16 xT[32 * 264];   // [n][c], padded stride

    // stage x^T tile (fp32 -> bf16)
    for (int idx = t; idx < 32 * 256; idx += 256) {
        const int n = idx & 31, c = idx >> 5;
        xT[n * 264 + c] =
            (__bf16)x[((size_t)(b * C_CH + c)) * N_TOK + nbase + n];
    }
    __syncthreads();

    const int msub = w & 1;        // which 16-row half of the 32-row tile
    const int jg   = w >> 1;       // 4 groups x 5 j-tiles = 20 tiles (320 cols)

    v8f acc[5];
#pragma unroll
    for (int u = 0; u < 5; ++u) {
        const int jbase = (jg * 5 + u) * 16;
        const int j     = jbase + lo;
        const float bias =
            (jbase < 32) ? bq[j] : (jbase < 64) ? bk[j - 32] : bv[j - 64];
#pragma unroll
        for (int r = 0; r < 8; ++r) acc[u][r] = bias;
    }

#pragma unroll 1
    for (int kk = 0; kk < C_CH; kk += 32) {
        // A fragment: row = lo, K runs [kk+8*hi, +8) and [kk+16+8*hi, +8)
        const __bf16* arow = &xT[(msub * 16 + lo) * 264 + kk + hi * 8];
        const v16bf A = bf_combine(*(const v8bf*)arow, *(const v8bf*)(arow + 16));

#pragma unroll
        for (int u = 0; u < 5; ++u) {
            const int jbase = (jg * 5 + u) * 16;
            const int j     = jbase + lo;
            const float* wrow = (jbase < 32) ? (wq + j * C_CH)
                              : (jbase < 64) ? (wk + (j - 32) * C_CH)
                                             : (wv + (j - 64) * C_CH);
            const float* p = wrow + kk + hi * 8;  // K runs [8*hi,+8) and [16+8*hi,+8)
            v16bf B;
#pragma unroll
            for (int e = 0; e < 16; ++e) B[e] = (__bf16)p[e < 8 ? e : e + 8];
            acc[u] = wmma_bf16(A, B, acc[u]);
        }
    }

    // store D tiles: row = r + 8*hi, col = jbase + lo
#pragma unroll
    for (int u = 0; u < 5; ++u) {
        const int jbase = (jg * 5 + u) * 16;
        const int j     = jbase + lo;
#pragma unroll
        for (int r = 0; r < 8; ++r) {
            const int n = nbase + msub * 16 + r + hi * 8;
            const __bf16 val = (__bf16)acc[u][r];
            if (jbase < 32)
                qT[((size_t)(b * N_TOK + n)) * D_CH + j] = val;
            else if (jbase < 64)
                kT[((size_t)(b * N_TOK + n)) * D_CH + (j - 32)] = val;
            else
                vN[((size_t)(b * C_CH + (j - 64))) * N_TOK + n] = val;
        }
    }
}

// ---------------------------------------------------------------------------
// Kernel 2: fused flash-style attention + output projection back into x-space.
// Grid: B * (N/16) blocks, 256 threads. Wave w handles keys [w*512, w*512+512),
// runs a private two-pass softmax, accumulates O[16 x 256] in registers,
// then waves merge with log-sum-exp rescaling through LDS atomics.
// ---------------------------------------------------------------------------
__global__ __launch_bounds__(256) void attn_kernel(
    const float* __restrict__ x, const float* __restrict__ gamma,
    const __bf16* __restrict__ qT, const __bf16* __restrict__ kT,
    const __bf16* __restrict__ vN, float* __restrict__ out)
{
    const int b     = blockIdx.x >> 8;
    const int mbase = (blockIdx.x & 255) * 16;
    const int t     = threadIdx.x;
    const int lane  = t & 31;
    const int w     = t >> 5;
    const int hi    = lane >> 4;
    const int lo    = lane & 15;

    __shared__ __align__(16) float  Sbuf[8][512];   // per-wave S bounce (16x32)
    __shared__ __align__(16) __bf16 Pbuf[8][512];   // per-wave P tile (16x32)
    __shared__ float wmaxL[8][16], wsumL[8][16], gm[16], dn[16];
    __shared__ float Obuf[16 * 256];

#pragma unroll
    for (int i = 0; i < 16; ++i) Obuf[t * 16 + i] = 0.0f;

    // Q fragment (A): row = mbase+lo, K runs d=[8*hi,+8) and [16+8*hi,+8)
    const __bf16* qrow = qT + ((size_t)(b * N_TOK + mbase + lo)) * D_CH + hi * 8;
    const v16bf Qf = bf_combine(*(const v8bf*)qrow, *(const v8bf*)(qrow + 16));

    const int kstart = w * 512;
    float mx = -3.0e38f, sm = 0.0f;   // per-row stats (valid in lanes 0..15)

    // ------------------------- pass 1: softmax stats -------------------------
#pragma unroll 1
    for (int ch = 0; ch < 16; ++ch) {
        const int nb = kstart + ch * 32;
        const __bf16* kr =
            kT + ((size_t)(b * N_TOK + nb + lo)) * D_CH + hi * 8;
        const v16bf B0 = bf_combine(*(const v8bf*)kr, *(const v8bf*)(kr + 16));
        const v16bf B1 = bf_combine(*(const v8bf*)(kr + 512),
                                    *(const v8bf*)(kr + 512 + 16));
        v8f S0 = {}, S1 = {};
        S0 = wmma_bf16(Qf, B0, S0);
        S1 = wmma_bf16(Qf, B1, S1);
#pragma unroll
        for (int r = 0; r < 8; ++r) {
            const int row = r + hi * 8;
            Sbuf[w][row * 32 + lo]      = S0[r];
            Sbuf[w][row * 32 + 16 + lo] = S1[r];
        }
        lds_fence();
        if (lane < 16) {
            float cm = -3.0e38f;
#pragma unroll
            for (int i = 0; i < 32; ++i) cm = fmaxf(cm, Sbuf[w][lane * 32 + i]);
            const float nm = fmaxf(mx, cm);
            sm *= __expf(mx - nm);
#pragma unroll
            for (int i = 0; i < 32; ++i) sm += __expf(Sbuf[w][lane * 32 + i] - nm);
            mx = nm;
        }
    }

    // --------------------- pass 2: P = exp(S-m), O += P*V --------------------
    v8f acc[16];
#pragma unroll
    for (int ct = 0; ct < 16; ++ct) acc[ct] = (v8f){};

#pragma unroll 1
    for (int ch = 0; ch < 16; ++ch) {
        const int nb = kstart + ch * 32;
        const __bf16* kr =
            kT + ((size_t)(b * N_TOK + nb + lo)) * D_CH + hi * 8;
        const v16bf B0 = bf_combine(*(const v8bf*)kr, *(const v8bf*)(kr + 16));
        const v16bf B1 = bf_combine(*(const v8bf*)(kr + 512),
                                    *(const v8bf*)(kr + 512 + 16));
        v8f S0 = {}, S1 = {};
        S0 = wmma_bf16(Qf, B0, S0);
        S1 = wmma_bf16(Qf, B1, S1);
#pragma unroll
        for (int r = 0; r < 8; ++r) {
            const int row = r + hi * 8;
            Sbuf[w][row * 32 + lo]      = S0[r];
            Sbuf[w][row * 32 + 16 + lo] = S1[r];
        }
        lds_fence();
        if (lane < 16) {
#pragma unroll
            for (int i = 0; i < 32; ++i)
                Pbuf[w][lane * 32 + i] =
                    (__bf16)__expf(Sbuf[w][lane * 32 + i] - mx);
        }
        lds_fence();
        // A fragment from P tile
        const __bf16* pr = &Pbuf[w][lo * 32 + hi * 8];
        const v16bf Af = bf_combine(*(const v8bf*)pr, *(const v8bf*)(pr + 16));
#pragma unroll
        for (int ct = 0; ct < 16; ++ct) {
            const __bf16* vr =
                vN + ((size_t)(b * C_CH + ct * 16 + lo)) * N_TOK + nb + hi * 8;
            const v16bf Bf =
                bf_combine(*(const v8bf*)vr, *(const v8bf*)(vr + 16));
            acc[ct] = wmma_bf16(Af, Bf, acc[ct]);
        }
    }

    // --------------------------- cross-wave merge ----------------------------
    if (lane < 16) { wmaxL[w][lane] = mx; wsumL[w][lane] = sm; }
    __syncthreads();
    if (w == 0 && lane < 16) {
        float g = -3.0e38f;
#pragma unroll
        for (int ww = 0; ww < 8; ++ww) g = fmaxf(g, wmaxL[ww][lane]);
        float d = 0.0f;
#pragma unroll
        for (int ww = 0; ww < 8; ++ww)
            d += __expf(wmaxL[ww][lane] - g) * wsumL[ww][lane];
        gm[lane] = g; dn[lane] = d;
    }
    __syncthreads();

#pragma unroll
    for (int ct = 0; ct < 16; ++ct) {
#pragma unroll
        for (int r = 0; r < 8; ++r) {
            const int row = r + hi * 8;
            const float sc = __expf(wmaxL[w][row] - gm[row]);
            atomicAdd(&Obuf[row * 256 + ct * 16 + lo], acc[ct][r] * sc);
        }
    }
    __syncthreads();

    // ------------------------ epilogue: gamma*O + x --------------------------
    const float g = gamma[0];
    const int m  = t & 15;
    const int c0 = t >> 4;
#pragma unroll
    for (int i = 0; i < 16; ++i) {
        const int c = c0 + 16 * i;
        const size_t gi = ((size_t)(b * C_CH + c)) * N_TOK + mbase + m;
        out[gi] = g * (Obuf[m * 256 + c] / dn[m]) + x[gi];
    }
}

// ---------------------------------------------------------------------------
extern "C" void kernel_launch(void* const* d_in, const int* in_sizes, int n_in,
                              void* d_out, int out_size, void* d_ws, size_t ws_size,
                              hipStream_t stream) {
    const float* x     = (const float*)d_in[0];
    const float* wq    = (const float*)d_in[1];
    const float* bq    = (const float*)d_in[2];
    const float* wk    = (const float*)d_in[3];
    const float* bk    = (const float*)d_in[4];
    const float* wv    = (const float*)d_in[5];
    const float* bv    = (const float*)d_in[6];
    const float* gamma = (const float*)d_in[7];
    float* out = (float*)d_out;

    char* ws = (char*)d_ws;
    __bf16* qT = (__bf16*)(ws);                         // 4*4096*32*2  = 1 MiB
    __bf16* kT = (__bf16*)(ws + (1u << 20));            // 1 MiB
    __bf16* vN = (__bf16*)(ws + (2u << 20));            // 4*256*4096*2 = 8 MiB

    qkv_proj_kernel<<<4 * (N_TOK / 32), 256, 0, stream>>>(
        x, wq, bq, wk, bk, wv, bv, qT, kT, vN);
    attn_kernel<<<4 * (N_TOK / 16), 256, 0, stream>>>(
        x, gamma, qT, kT, vN, out);
}
